// SequenceDecoder_4355096838348
// MI455X (gfx1250) — matrix operand order repaired and checked by reading, not compiled, not run on hardware
//
#include <hip/hip_runtime.h>
#include <hip/hip_bf16.h>
#include <stddef.h>

// Problem constants (match reference): B=256, S=512, H=512, L=2
#define BB   256
#define HH   512
#define SS   512
#define NT   96      // N tiles per packed weight matrix: 3H/16 = 1536/16
#define KTI  16      // K tiles: H/32 = 512/32
#define NWG  64      // persistent grid: 64 workgroups
#define TPB  256     // 8 waves of 32
#define WPM  (KTI * NT * 32 * 16)   // bf16 elements per packed weight matrix
#define LROW 520     // padded LDS row stride in bf16 (1040 B -> start bank = 4*lane)

typedef __attribute__((ext_vector_type(16))) __bf16 v16bf;
typedef __attribute__((ext_vector_type(8)))  __bf16 v8bf;
typedef __attribute__((ext_vector_type(8)))  float  v8f;

// ---------- helpers ----------
__device__ __forceinline__ unsigned short f2bf_bits(float f) {
    unsigned u = __float_as_uint(f);
    u = u + 0x7FFFu + ((u >> 16) & 1u);      // round-to-nearest-even
    return (unsigned short)(u >> 16);
}
__device__ __forceinline__ v8f wmma_bf16(v16bf a, v16bf b, v8f c) {
    // emits v_wmma_f32_16x16x32_bf16
    return __builtin_amdgcn_wmma_f32_16x16x32_bf16(false, a, false, b, (short)0, c, false, false);
}
__device__ __forceinline__ float fast_rcp(float x) { return __builtin_amdgcn_rcpf(x); }
__device__ __forceinline__ float sigmoidf_(float x) {
    return fast_rcp(1.0f + __expf(-x));                 // v_exp + v_rcp, no IEEE divide
}
__device__ __forceinline__ float tanhf_(float x) {
    x = fminf(fmaxf(x, -15.0f), 15.0f);
    float e = __expf(2.0f * x);
    return (e - 1.0f) * fast_rcp(e + 1.0f);
}

// A fragment (16x32 bf16) read from the LDS-staged tile: two ds_load_b128 per lane,
// zero conversion ALU. Layout: lanes 0-15 rows M=0..15; elems0..15 = K(kb..kb+15),
// kb = 16*(lane>>4) (matches packed B).
__device__ __forceinline__ v16bf lds_a_frag(const unsigned short* lds_tile, int kt, int lane) {
    int row = lane & 15;
    int kb  = kt * 32 + ((lane >> 4) << 4);
    const v8bf* p = (const v8bf*)(lds_tile + row * LROW + kb);
    v8bf lo = p[0];        // K kb .. kb+7
    v8bf hi = p[1];        // K kb+8 .. kb+15
    v16bf a;
#pragma unroll
    for (int i = 0; i < 8; ++i) { a[i] = lo[i]; a[8 + i] = hi[i]; }
    return a;
}

// B fragment: pre-packed, 16 contiguous bf16 (32B) per lane -> 2x global b128 loads.
__device__ __forceinline__ v16bf load_b_frag(const unsigned short* __restrict__ wp,
                                             int kt, int nt, int lane) {
    return *(const v16bf*)(wp + ((((size_t)kt * NT + nt) * 32 + lane) << 4));
}

// Cooperative copy of one 16x512 bf16 A-tile (rows contiguous in global) into padded LDS.
__device__ __forceinline__ void stage_tile(unsigned short* dst, const unsigned short* src) {
    const float4* gs = (const float4*)src;   // 16 rows x 64 float4, contiguous
    float4*       ls = (float4*)dst;         // row stride 65 float4 (1040 B)
    for (int c = threadIdx.x; c < 16 * 64; c += TPB) {
        int row = c >> 6, col = c & 63;
        ls[row * 65 + col] = gs[c];
    }
}

// Device-wide software barrier (arrive count + generation).
__device__ __forceinline__ void grid_barrier(unsigned* bar) {
    __threadfence();
    __syncthreads();
    if (threadIdx.x == 0) {
        unsigned g = __hip_atomic_load(&bar[1], __ATOMIC_RELAXED, __HIP_MEMORY_SCOPE_AGENT);
        unsigned prev = __hip_atomic_fetch_add(&bar[0], 1u, __ATOMIC_ACQ_REL, __HIP_MEMORY_SCOPE_AGENT);
        if (prev == NWG - 1) {
            __hip_atomic_store(&bar[0], 0u, __ATOMIC_RELAXED, __HIP_MEMORY_SCOPE_AGENT);
            __hip_atomic_fetch_add(&bar[1], 1u, __ATOMIC_RELEASE, __HIP_MEMORY_SCOPE_AGENT);
        } else {
            while (__hip_atomic_load(&bar[1], __ATOMIC_ACQUIRE, __HIP_MEMORY_SCOPE_AGENT) == g)
                __builtin_amdgcn_s_sleep(2);
        }
    }
    __syncthreads();
    __threadfence();
}

// ---------- setup kernels ----------
// Zero barrier state, x0 = 0, h0/h1 <- encoder_hidden (f32 master + bf16 mirror).
__global__ void seqdec_init(const float* __restrict__ eh,
                            float* h0buf, float* h1buf,
                            unsigned short* h0bf, unsigned short* h1bf,
                            float* xbuf, unsigned* bar) {
    int tid = blockIdx.x * blockDim.x + threadIdx.x;
    const int n = BB * HH;
    if (tid < n) {
        float a = eh[tid], b = eh[n + tid];
        h0buf[tid] = a;  h0bf[tid] = f2bf_bits(a);   // layer 0, parity-0 buffers
        h1buf[tid] = b;  h1bf[tid] = f2bf_bits(b);   // layer 1, parity-0 buffers
    }
    if (tid < BB) xbuf[tid] = 0.0f;
    if (tid < 2)  bar[tid] = 0u;
}

// Pack W_hh0 / W_ih1 / W_hh1 (each 1536x512 fp32, row=output) into bf16
// B-fragment order: value = W[n][k], n = nt*16+(lane&15), k = kt*32+16*(lane>>4)+i.
__global__ void seqdec_pack(const float* __restrict__ Whh0,
                            const float* __restrict__ Wih1,
                            const float* __restrict__ Whh1,
                            unsigned short* __restrict__ wp) {
    int tid = blockIdx.x * blockDim.x + threadIdx.x;
    const int total = 3 * KTI * NT * 32;
    if (tid >= total) return;
    int lane = tid & 31;
    int nt   = (tid >> 5) % NT;
    int kt   = ((tid >> 5) / NT) % KTI;
    int mat  = tid / (KTI * NT * 32);
    const float* W = (mat == 0) ? Whh0 : (mat == 1) ? Wih1 : Whh1;
    int n  = nt * 16 + (lane & 15);
    int kb = kt * 32 + ((lane >> 4) << 4);
    const float* src = W + (size_t)n * HH + kb;
    unsigned short* dst = wp + (size_t)mat * WPM + ((((size_t)kt * NT + nt) * 32 + lane) << 4);
#pragma unroll
    for (int i = 0; i < 16; ++i) dst[i] = f2bf_bits(src[i]);
}

// ---------- persistent decoder ----------
__global__ void __launch_bounds__(TPB)
seqdec_persistent(const float* __restrict__ Wih0, const float* __restrict__ bih0,
                  const float* __restrict__ bhh0, const float* __restrict__ bih1,
                  const float* __restrict__ bhh1, const float* __restrict__ Wo,
                  const float* __restrict__ bo,
                  const unsigned short* __restrict__ pWhh0,
                  const unsigned short* __restrict__ pWih1,
                  const unsigned short* __restrict__ pWhh1,
                  float* __restrict__ h0buf, float* __restrict__ h1buf,
                  unsigned short* __restrict__ h0bfb, unsigned short* __restrict__ h1bfb,
                  float* __restrict__ xbuf, unsigned* bar,
                  float* __restrict__ out) {
    __shared__ __align__(16) unsigned short ldsA[2][16 * LROW];   // 2 x 16.25 KB

    const int lane  = threadIdx.x & 31;
    const int wid   = threadIdx.x >> 5;
    const int gw    = blockIdx.x * (TPB / 32) + wid;   // 0..511
    const int mtile = gw >> 5;                         // 0..15 (constant per WG)
    const int jtile = gw & 31;                         // 0..31
    const int m8    = (lane >> 4) << 3;                // 0 or 8: D-frag row base
    const int jcol  = jtile * 16 + (lane & 15);        // hidden column 0..511
    const int brow  = mtile * 16 + m8;                 // batch-row base of D elems

    // Per-column constants (invariant across time steps).
    const float xw_r = Wih0[jcol],  xw_z = Wih0[HH + jcol],  xw_n = Wih0[2 * HH + jcol];
    const float bi0r = bih0[jcol],  bi0z = bih0[HH + jcol],  bi0n = bih0[2 * HH + jcol];
    const float bh0r = bhh0[jcol],  bh0z = bhh0[HH + jcol],  bh0n = bhh0[2 * HH + jcol];
    const float bi1r = bih1[jcol],  bi1z = bih1[HH + jcol],  bi1n = bih1[2 * HH + jcol];
    const float bh1r = bhh1[jcol],  bh1z = bhh1[HH + jcol],  bh1n = bhh1[2 * HH + jcol];

    for (int t = 0; t < SS; ++t) {
        const float*          h0r   = h0buf + (size_t)(t & 1) * BB * HH;
        float*                h0w   = h0buf + (size_t)((t + 1) & 1) * BB * HH;
        const float*          h1r   = h1buf + (size_t)(t & 1) * BB * HH;
        float*                h1w   = h1buf + (size_t)((t + 1) & 1) * BB * HH;
        const unsigned short* h0rbf = h0bfb + (size_t)(t & 1) * BB * HH;
        unsigned short*       h0wbf = h0bfb + (size_t)((t + 1) & 1) * BB * HH;
        const unsigned short* h1rbf = h1bfb + (size_t)(t & 1) * BB * HH;
        unsigned short*       h1wbf = h1bfb + (size_t)((t + 1) & 1) * BB * HH;

        // ---- Phase 1: stage A-tiles once per WG, then gh0 = h0 @ Whh0^T and
        //      gh1 = h1 @ Whh1^T (3 gate tiles each); fuse layer-0 GRU -> h0_new.
        //      gh1 accumulators stay in VGPRs across the barrier. ----
        stage_tile(&ldsA[0][0], h0rbf + (size_t)mtile * 16 * HH);
        stage_tile(&ldsA[1][0], h1rbf + (size_t)mtile * 16 * HH);
        __syncthreads();

        v8f r0 = {0,0,0,0,0,0,0,0}, z0 = r0, n0 = r0;
        v8f r1 = r0, z1 = r0, n1 = r0;
        for (int kt = 0; kt < KTI; ++kt) {
            v16bf A0 = lds_a_frag(&ldsA[0][0], kt, lane);
            v16bf A1 = lds_a_frag(&ldsA[1][0], kt, lane);
            r0 = wmma_bf16(A0, load_b_frag(pWhh0, kt, jtile,      lane), r0);
            z0 = wmma_bf16(A0, load_b_frag(pWhh0, kt, jtile + 32, lane), z0);
            n0 = wmma_bf16(A0, load_b_frag(pWhh0, kt, jtile + 64, lane), n0);
            r1 = wmma_bf16(A1, load_b_frag(pWhh1, kt, jtile,      lane), r1);
            z1 = wmma_bf16(A1, load_b_frag(pWhh1, kt, jtile + 32, lane), z1);
            n1 = wmma_bf16(A1, load_b_frag(pWhh1, kt, jtile + 64, lane), n1);
        }
#pragma unroll
        for (int i = 0; i < 8; ++i) {
            int b = brow + i;
            float x  = xbuf[b];
            float rr = sigmoidf_(x * xw_r + bi0r + r0[i] + bh0r);
            float zz = sigmoidf_(x * xw_z + bi0z + z0[i] + bh0z);
            float nn = tanhf_(x * xw_n + bi0n + rr * (n0[i] + bh0n));
            float hn = (1.0f - zz) * nn + zz * h0r[(size_t)b * HH + jcol];
            h0w[(size_t)b * HH + jcol]   = hn;
            h0wbf[(size_t)b * HH + jcol] = f2bf_bits(hn);
        }
        grid_barrier(bar);

        // ---- Phase 2: gi1 = h0_new @ Wih1^T (3 gate tiles); fuse layer-1 GRU -> h1_new ----
        stage_tile(&ldsA[0][0], h0wbf + (size_t)mtile * 16 * HH);
        __syncthreads();

        v8f cr = {0,0,0,0,0,0,0,0}, cz = cr, cn = cr;
        for (int kt = 0; kt < KTI; ++kt) {
            v16bf A = lds_a_frag(&ldsA[0][0], kt, lane);
            cr = wmma_bf16(A, load_b_frag(pWih1, kt, jtile,      lane), cr);
            cz = wmma_bf16(A, load_b_frag(pWih1, kt, jtile + 32, lane), cz);
            cn = wmma_bf16(A, load_b_frag(pWih1, kt, jtile + 64, lane), cn);
        }
#pragma unroll
        for (int i = 0; i < 8; ++i) {
            int b = brow + i;
            float rr = sigmoidf_(cr[i] + bi1r + r1[i] + bh1r);
            float zz = sigmoidf_(cz[i] + bi1z + z1[i] + bh1z);
            float nn = tanhf_(cn[i] + bi1n + rr * (n1[i] + bh1n));
            float hn = (1.0f - zz) * nn + zz * h1r[(size_t)b * HH + jcol];
            h1w[(size_t)b * HH + jcol]   = hn;
            h1wbf[(size_t)b * HH + jcol] = f2bf_bits(hn);
        }
        grid_barrier(bar);

        // ---- Phase 3: y[b] = h1_new[b,:] . Wo + bo ; y feeds next step's input ----
        if (gw < BB) {
            const float* hr = h1w + (size_t)gw * HH;
            int k0 = lane * 16;
            float s = 0.0f;
#pragma unroll
            for (int i = 0; i < 16; ++i) s += hr[k0 + i] * Wo[k0 + i];
#pragma unroll
            for (int off = 16; off > 0; off >>= 1) s += __shfl_xor(s, off, 32);
            if (lane == 0) {
                float y = s + bo[0];
                out[(size_t)gw * SS + t] = y;   // decoder_outputs (B,S,1)
                xbuf[gw] = y;                   // feed_previous
            }
        }
        grid_barrier(bar);
    }

    // ---- Tail: decoder_hidden = stack([h0_final, h1_final]) at out + B*S ----
    const float* h0f = h0buf + (size_t)(SS & 1) * BB * HH;
    const float* h1f = h1buf + (size_t)(SS & 1) * BB * HH;
    float* dh = out + (size_t)BB * SS;
    for (int idx = blockIdx.x * TPB + threadIdx.x; idx < BB * HH; idx += NWG * TPB) {
        dh[idx] = h0f[idx];
        dh[BB * HH + idx] = h1f[idx];
    }
}

// ---------- launch ----------
extern "C" void kernel_launch(void* const* d_in, const int* in_sizes, int n_in,
                              void* d_out, int out_size, void* d_ws, size_t ws_size,
                              hipStream_t stream) {
    (void)in_sizes; (void)n_in; (void)out_size; (void)ws_size;
    const float* eh   = (const float*)d_in[1];   // encoder_hidden (2,B,H)
    const float* Wih0 = (const float*)d_in[2];
    const float* Whh0 = (const float*)d_in[3];
    const float* bih0 = (const float*)d_in[4];
    const float* bhh0 = (const float*)d_in[5];
    const float* Wih1 = (const float*)d_in[6];
    const float* Whh1 = (const float*)d_in[7];
    const float* bih1 = (const float*)d_in[8];
    const float* bhh1 = (const float*)d_in[9];
    const float* Wo   = (const float*)d_in[10];
    const float* bo   = (const float*)d_in[11];
    float* out = (float*)d_out;

    char* ws = (char*)d_ws;
    float*          h0buf = (float*)(ws + 0x000000);            // 2 x 256x512 f32 (1 MB)
    float*          h1buf = (float*)(ws + 0x100000);            // 2 x 256x512 f32 (1 MB)
    float*          xbuf  = (float*)(ws + 0x200000);            // 256 f32
    unsigned*       bar   = (unsigned*)(ws + 0x201000);         // {count, gen}
    unsigned short* wpack = (unsigned short*)(ws + 0x202000);   // 3 x 1.5 MB bf16
    unsigned short* h0bf  = (unsigned short*)(ws + 0x690000);   // 2 x 256 KB bf16 mirror
    unsigned short* h1bf  = (unsigned short*)(ws + 0x710000);   // 2 x 256 KB bf16 mirror

    seqdec_init<<<(BB * HH + 255) / 256, 256, 0, stream>>>(eh, h0buf, h1buf, h0bf, h1bf, xbuf, bar);
    seqdec_pack<<<(3 * KTI * NT * 32 + 255) / 256, 256, 0, stream>>>(Whh0, Wih1, Whh1, wpack);
    seqdec_persistent<<<NWG, TPB, 0, stream>>>(
        Wih0, bih0, bhh0, bih1, bhh1, Wo, bo,
        wpack, wpack + WPM, wpack + 2 * (size_t)WPM,
        h0buf, h1buf, h0bf, h1bf, xbuf, bar, out);
}
